// SelfAttention_75703093559944
// MI455X (gfx1250) — compile-verified, matrix operand order checked by reading, not run on hardware
//
#include <hip/hip_runtime.h>
#include <hip/hip_bf16.h>
#include <cmath>

typedef __attribute__((ext_vector_type(16))) _Float16 v16h;
typedef __attribute__((ext_vector_type(8)))  float    v8f;

#define S_LEN 2048
#define BATCH 8
#define DMODEL 1024
#define HD 64

// ---------------------------------------------------------------------------
// WMMA helper: D = A(16x32 f16) * B(32x16 f16) + C(16x16 f32)
// ---------------------------------------------------------------------------
__device__ __forceinline__ v8f wmma_f16(v16h a, v16h b, v8f c) {
  return __builtin_amdgcn_wmma_f32_16x16x32_f16(
      /*neg_a=*/false, a, /*neg_b=*/false, b,
      /*c_mod=*/(short)0, c, /*reuse_a=*/false, /*reuse_b=*/false);
}

// ---------------------------------------------------------------------------
// Fragment loaders.
// A 16x32 f16 layout (ISA 7.12.2): lane L -> m=L&15, h=L>>4.
//   VGPR j in [0,3]: K = 8h + 2j, 2j+1 ; VGPR j in [4,7]: K = 16 + 8h + 2(j-4)
// B mirrors A on the transposed operand (row index = N column, k pattern same).
// Contiguous K pairs -> dword loads.
// ---------------------------------------------------------------------------
union FragBits { unsigned int u[8]; v16h v; };

__device__ __forceinline__ v16h load_frag_h(const _Float16* p, int row, int h, int ld) {
  const unsigned int* q = (const unsigned int*)(p + row * ld + 8 * h);
  FragBits t;
#pragma unroll
  for (int j = 0; j < 8; ++j) t.u[j] = q[(j < 4) ? j : j + 4];
  return t.v;
}

__device__ __forceinline__ v16h load_frag_f32(const float* tile, int m, int h, int ld) {
  const float* row = tile + m * ld + 8 * h;
  v16h a;
#pragma unroll
  for (int j = 0; j < 8; ++j) {
    int k = (j < 4) ? 2 * j : 16 + 2 * (j - 4);
    float2 f = *(const float2*)(row + k);
    a[2 * j]     = (_Float16)f.x;
    a[2 * j + 1] = (_Float16)f.y;
  }
  return a;
}

// ---------------------------------------------------------------------------
// Kernel 0: convert weights fp32 [k][n] -> f16 transposed [w][n][k]
// ---------------------------------------------------------------------------
__global__ void __launch_bounds__(256)
prep_weights(const float* __restrict__ Wq, const float* __restrict__ Wk,
             const float* __restrict__ Wv, _Float16* __restrict__ Wt) {
  int idx = blockIdx.x * 256 + threadIdx.x;          // 0 .. 3*64*1024-1
  int w   = idx >> 16;                               // 64*1024 per matrix
  int rem = idx & 65535;
  int n   = rem >> 10;
  int k   = rem & 1023;
  const float* W = (w == 0) ? Wq : ((w == 1) ? Wk : Wv);
  Wt[idx] = (_Float16)W[k * HD + n];
}

// ---------------------------------------------------------------------------
// Kernel 1: QKV projection. One wave computes a 16-row x 192-col tile.
//   X      : [S*B, 1024] fp32 (row r = s*B + b)
//   Wt     : [3][64][1024] f16 (transposed weights)
//   Qh,Kh  : [b][s][64] f16
//   Vt     : [b][d][2048] f16 (transposed V for PV B-operand)
// ---------------------------------------------------------------------------
__global__ void __launch_bounds__(32)
qkv_proj(const float* __restrict__ X, const _Float16* __restrict__ Wt,
         const float* __restrict__ bq, const float* __restrict__ bk,
         const float* __restrict__ bv,
         _Float16* __restrict__ Qh, _Float16* __restrict__ Kh,
         _Float16* __restrict__ Vt) {
  const int lane = threadIdx.x;
  const int m = lane & 15, h = lane >> 4;
  const int rt = blockIdx.x;                 // 16-row tile index (0..1023)

  v8f acc[12];
#pragma unroll
  for (int t = 0; t < 12; ++t)
#pragma unroll
    for (int i = 0; i < 8; ++i) acc[t][i] = 0.0f;

  const float* Xtile = X + (size_t)rt * 16 * DMODEL;
  for (int k0 = 0; k0 < DMODEL; k0 += 32) {
    v16h a = load_frag_f32(Xtile + k0, m, h, DMODEL);
#pragma unroll
    for (int w = 0; w < 3; ++w)
#pragma unroll
      for (int nt = 0; nt < 4; ++nt) {
        const _Float16* Wb = Wt + (size_t)(w * 64 + nt * 16) * DMODEL + k0;
        v16h bf = load_frag_h(Wb, m, h, DMODEL);
        acc[w * 4 + nt] = wmma_f16(a, bf, acc[w * 4 + nt]);
      }
  }

  // epilogue: bias add + scatter into Qh / Kh / Vt
#pragma unroll
  for (int w = 0; w < 3; ++w) {
    const float* bias = (w == 0) ? bq : ((w == 1) ? bk : bv);
#pragma unroll
    for (int nt = 0; nt < 4; ++nt) {
      const int col = nt * 16 + m;
      const float bv_ = bias[col];
      v8f c = acc[w * 4 + nt];
#pragma unroll
      for (int i = 0; i < 8; ++i) {
        int r = rt * 16 + i + 8 * h;
        int s = r >> 3, b = r & 7;
        _Float16 val = (_Float16)(c[i] + bv_);
        if (w == 0)      Qh[((size_t)b * S_LEN + s) * HD + col] = val;
        else if (w == 1) Kh[((size_t)b * S_LEN + s) * HD + col] = val;
        else             Vt[(size_t)b * HD * S_LEN + (size_t)col * S_LEN + s] = val;
      }
    }
  }
}

// ---------------------------------------------------------------------------
// Kernel 2: flash attention. 4 waves/block, each wave owns 16 query rows.
// Score C-layout: lane L holds (N = L&15, M = i + 8*(L>>4)) for i=0..7, so a
// score row lives in one VGPR across a 16-lane half -> shfl_xor {1,2,4,8}
// reductions for online softmax.
// ---------------------------------------------------------------------------
__global__ void __launch_bounds__(128)
flash_attn(const _Float16* __restrict__ Qh, const _Float16* __restrict__ Kh,
           const _Float16* __restrict__ Vt, const float* __restrict__ mask,
           float* __restrict__ out) {
  __shared__ __align__(16) _Float16 Plds[4 * 16 * 32];

  const int lane = threadIdx.x & 31;
  const int wid  = threadIdx.x >> 5;
  const int n = lane & 15, h = lane >> 4;
  const int b  = blockIdx.y;
  const int qt = blockIdx.x * 4 + wid;             // 16-row query tile

  const _Float16* Qbase = Qh + ((size_t)b * S_LEN + qt * 16) * HD;
  const v16h aq0 = load_frag_h(Qbase,      n, h, HD);
  const v16h aq1 = load_frag_h(Qbase + 32, n, h, HD);

  float mrun[8], lrun[8];
  v8f O[4];
#pragma unroll
  for (int i = 0; i < 8; ++i) { mrun[i] = -1e30f; lrun[i] = 0.0f; }
#pragma unroll
  for (int t = 0; t < 4; ++t)
#pragma unroll
    for (int i = 0; i < 8; ++i) O[t][i] = 0.0f;

  const float scale = 0.03125f;                    // 1/sqrt(1024)
  _Float16* Pw = Plds + wid * 512;
  const _Float16* Vbb = Vt + (size_t)b * HD * S_LEN;

  for (int kb = 0; kb < S_LEN; kb += 32) {
    // ---- scores for 32 keys: two 16x16 tiles, K-dim 64 = 2 wmma each ----
    const _Float16* Kb0 = Kh + ((size_t)b * S_LEN + kb) * HD;
    __builtin_prefetch(Kb0 + 32 * HD, 0, 1);       // next key block
    __builtin_prefetch(Vbb + kb + 32, 0, 1);
    v8f c0, c1;
#pragma unroll
    for (int i = 0; i < 8; ++i) { c0[i] = 0.0f; c1[i] = 0.0f; }
    c0 = wmma_f16(aq0, load_frag_h(Kb0,            n, h, HD), c0);
    c0 = wmma_f16(aq1, load_frag_h(Kb0 + 32,       n, h, HD), c0);
    c1 = wmma_f16(aq0, load_frag_h(Kb0 + 16 * HD,      n, h, HD), c1);
    c1 = wmma_f16(aq1, load_frag_h(Kb0 + 16 * HD + 32, n, h, HD), c1);

    // ---- (score + log(mask)) * scale, online softmax ----
    const float lm0 = __logf(mask[(kb + n) * BATCH + b]);
    const float lm1 = __logf(mask[(kb + 16 + n) * BATCH + b]);
    float s0[8], s1[8], rmax[8];
#pragma unroll
    for (int i = 0; i < 8; ++i) {
      s0[i] = (c0[i] + lm0) * scale;
      s1[i] = (c1[i] + lm1) * scale;
      rmax[i] = fmaxf(s0[i], s1[i]);
    }
#pragma unroll
    for (int xm = 1; xm <= 8; xm <<= 1)
#pragma unroll
      for (int i = 0; i < 8; ++i)
        rmax[i] = fmaxf(rmax[i], __shfl_xor(rmax[i], xm, 32));

    float alpha[8], p0[8], p1[8], rsum[8];
#pragma unroll
    for (int i = 0; i < 8; ++i) {
      float mn = fmaxf(mrun[i], rmax[i]);
      alpha[i] = __expf(mrun[i] - mn);
      mrun[i]  = mn;
      p0[i] = __expf(s0[i] - mn);
      p1[i] = __expf(s1[i] - mn);
      rsum[i] = p0[i] + p1[i];
    }
#pragma unroll
    for (int xm = 1; xm <= 8; xm <<= 1)
#pragma unroll
      for (int i = 0; i < 8; ++i)
        rsum[i] += __shfl_xor(rsum[i], xm, 32);
#pragma unroll
    for (int i = 0; i < 8; ++i) lrun[i] = lrun[i] * alpha[i] + rsum[i];
#pragma unroll
    for (int t = 0; t < 4; ++t)
#pragma unroll
      for (int i = 0; i < 8; ++i) O[t][i] *= alpha[i];

    // ---- reshape P (C-layout f32) -> A-fragment f16 via LDS ----
#pragma unroll
    for (int i = 0; i < 8; ++i) {
      int row = i + 8 * h;
      Pw[row * 32 + n]      = (_Float16)p0[i];
      Pw[row * 32 + 16 + n] = (_Float16)p1[i];
    }
    asm volatile("s_wait_dscnt 0" ::: "memory");
    v16h pa = load_frag_h(Pw, n, h, 32);

    // ---- O += P(16x32) x V(32x64) ----
#pragma unroll
    for (int t = 0; t < 4; ++t) {
      const _Float16* Vb = Vbb + (size_t)(t * 16) * S_LEN + kb;
      O[t] = wmma_f16(pa, load_frag_h(Vb, n, h, S_LEN), O[t]);
    }
  }

  // ---- normalize and write out [s][b][d] fp32 ----
  float inv[8];
#pragma unroll
  for (int i = 0; i < 8; ++i) inv[i] = 1.0f / lrun[i];
#pragma unroll
  for (int t = 0; t < 4; ++t)
#pragma unroll
    for (int i = 0; i < 8; ++i) {
      int s = qt * 16 + i + 8 * h;
      int d = t * 16 + n;
      out[((size_t)s * BATCH + b) * HD + d] = O[t][i] * inv[i];
    }
}

// ---------------------------------------------------------------------------
extern "C" void kernel_launch(void* const* d_in, const int* in_sizes, int n_in,
                              void* d_out, int out_size, void* d_ws, size_t ws_size,
                              hipStream_t stream) {
  const float* X    = (const float*)d_in[0];
  const float* mask = (const float*)d_in[1];
  const float* Wq   = (const float*)d_in[2];
  const float* bq   = (const float*)d_in[3];
  const float* Wk   = (const float*)d_in[4];
  const float* bk   = (const float*)d_in[5];
  const float* Wv   = (const float*)d_in[6];
  const float* bv   = (const float*)d_in[7];
  float* out = (float*)d_out;

  char* ws = (char*)d_ws;
  const size_t WT_BYTES  = (size_t)3 * 64 * 1024 * 2;          // 384 KB
  const size_t QKV_BYTES = (size_t)BATCH * S_LEN * HD * 2;     // 2 MB each
  _Float16* Wt = (_Float16*)ws;
  _Float16* Qh = (_Float16*)(ws + WT_BYTES);
  _Float16* Kh = (_Float16*)(ws + WT_BYTES + QKV_BYTES);
  _Float16* Vt = (_Float16*)(ws + WT_BYTES + 2 * QKV_BYTES);

  prep_weights<<<768, 256, 0, stream>>>(Wq, Wk, Wv, Wt);
  qkv_proj<<<(S_LEN * BATCH) / 16, 32, 0, stream>>>(X, Wt, bq, bk, bv, Qh, Kh, Vt);
  flash_attn<<<dim3(S_LEN / 64, BATCH), 128, 0, stream>>>(Qh, Kh, Vt, mask, out);
}
